// BidirLSTMLayer_2972117369277
// MI455X (gfx1250) — matrix operand order repaired and checked by reading, not compiled, not run on hardware
//
#include <hip/hip_runtime.h>

typedef __attribute__((ext_vector_type(16))) __bf16 v16bf;
typedef __attribute__((ext_vector_type(8)))  float  v8f;

#define T_ 512
#define B_ 64
#define I_ 512
#define H_ 512
#define NWG_DIR 8        // workgroups per direction (few -> cheap grid sync)
#define WG_THREADS 512   // 16 waves of 32

__device__ __forceinline__ unsigned short f32_to_bf16(float f) {
  unsigned u = __float_as_uint(f);
  unsigned r = 0x7FFFu + ((u >> 16) & 1u);   // round-to-nearest-even
  return (unsigned short)((u + r) >> 16);
}

__device__ __forceinline__ float sigmoidf_(float x) {
  return 1.0f / (1.0f + __expf(-x));
}

union FragA { unsigned int u32[8]; v16bf v; };

// ---------------- prep kernels ----------------

// x f32 -> bf16 (streamed once; 32 MB total)
__global__ void k_cvt_x(const float* __restrict__ x, unsigned short* __restrict__ xb, int n) {
  int i = blockIdx.x * blockDim.x + threadIdx.x;
  if (i < n) xb[i] = f32_to_bf16(x[i]);
}

// Pack [Wih | Whh] (per direction, 4H x (I+H)) into WMMA B-fragment order:
// layout [dir][ct(128)][kt(32)][lane(32)][e(16)] bf16.
// B 32x16 16-bit layout: K = kt*32 + e + 8*(lane>=16) + 8*(e>=8),
// N = ct*16 + (lane & 15).
__global__ void k_pack_w(const float* __restrict__ Wih_f, const float* __restrict__ Whh_f,
                         const float* __restrict__ Wih_b, const float* __restrict__ Whh_b,
                         unsigned short* __restrict__ wp) {
  int idx = blockIdx.x * blockDim.x + threadIdx.x;
  const int n = 2 * 128 * 32 * 512;
  if (idx >= n) return;
  int e    = idx & 15;
  int lane = (idx >> 4) & 31;
  int kt   = (idx >> 9) & 31;
  int ct   = (idx >> 14) & 127;
  int dir  = idx >> 21;
  int K    = kt * 32 + e + 8 * (lane >> 4) + 8 * (e >> 3);
  int col  = ct * 16 + (lane & 15);           // gate column 0..2047
  const float* Wih = dir ? Wih_b : Wih_f;
  const float* Whh = dir ? Whh_b : Whh_f;
  float val = (K < I_) ? Wih[(size_t)col * I_ + K] : Whh[(size_t)col * H_ + (K - I_)];
  wp[idx] = f32_to_bf16(val);
}

// init h ping-pong buffer (parity 0) from h0 for both directions; zero sync counters
__global__ void k_init_h(const float* __restrict__ h0f, const float* __restrict__ h0b,
                         unsigned short* __restrict__ hbuf, unsigned int* __restrict__ cnt) {
  int i = blockIdx.x * blockDim.x + threadIdx.x;
  const int n = B_ * H_;
  if (i < n) {
    hbuf[0 * n + i] = f32_to_bf16(h0f[i]);   // parity 0, dir 0
    hbuf[1 * n + i] = f32_to_bf16(h0b[i]);   // parity 0, dir 1
  }
  if (i < 2) cnt[i] = 0;
}

// ---------------- persistent fused scan ----------------
// grid = 2 * NWG_DIR blocks. Block owns 64 hidden units of one direction.
// Per step: gates = [x_t | h] @ Wcomb^T + b via bf16 WMMA; c in registers;
// h shared through a double-buffered bf16 buffer with per-direction grid sync.
__global__ __launch_bounds__(WG_THREADS)
void k_scan(const unsigned short* __restrict__ xb,
            const unsigned short* __restrict__ wp,
            unsigned short* __restrict__ hbuf,
            unsigned int* __restrict__ cnt,
            const float* __restrict__ c0f, const float* __restrict__ c0b,
            const float* __restrict__ bfw, const float* __restrict__ bbw,
            float* __restrict__ out)
{
  const int dir  = blockIdx.x / NWG_DIR;
  const int wg   = blockIdx.x % NWG_DIR;
  const int tid  = threadIdx.x;
  const int wave = tid >> 5;
  const int lane = tid & 31;
  const int mr   = wave >> 2;      // row tile 0..3 (B = 64)
  const int wcol = wave & 3;       // hidden col tile within WG
  const int l16  = lane & 15;
  const int hi   = lane >> 4;

  const int hcol = wg * 64 + wcol * 16 + l16;  // hidden unit 0..511 (D/C: N = lane%16)
  const int hct  = wg * 4 + wcol;              // hidden col-tile 0..31
  const int mA   = mr * 16 + l16;              // A-frag row (lanes 0..15 / 16..31 both M=l16)

  const float* c0   = dir ? c0b : c0f;
  const float* bias = dir ? bbw : bfw;
  const unsigned short* wpd = wp + (size_t)dir * (128u * 32u * 512u);

  // A-frag per-VGPR K offsets (16-bit A 16x32 layout)
  int kb[8];
#pragma unroll
  for (int v = 0; v < 8; ++v)
    kb[v] = ((v < 4) ? (2 * v) : (16 + 2 * (v - 4))) + 8 * hi;

  const float bI = bias[0 * H_ + hcol];
  const float bF = bias[1 * H_ + hcol];
  const float bG = bias[2 * H_ + hcol];
  const float bO = bias[3 * H_ + hcol];

  float c[8];
#pragma unroll
  for (int e = 0; e < 8; ++e)
    c[e] = c0[(size_t)(mr * 16 + e + 8 * hi) * H_ + hcol];

  unsigned int* mycnt = cnt + dir;
  const int n = B_ * H_;

  for (int s = 0; s < T_; ++s) {
    const int t = dir ? (T_ - 1 - s) : s;   // bwd consumes flipped x, writes flipped-back out
    const unsigned short* hread  = hbuf + ((size_t)(s & 1) * 2 + dir) * n;
    unsigned short*       hwrite = hbuf + ((size_t)((s + 1) & 1) * 2 + dir) * n;

    v8f ai, af2, ag, ao;
#pragma unroll
    for (int e = 0; e < 8; ++e) { ai[e] = 0.f; af2[e] = 0.f; ag[e] = 0.f; ao[e] = 0.f; }

#define BFRAG(g, kt) (*(const v16bf*)(wpd + (((size_t)((g) * 32 + hct) * 32 + (size_t)(kt)) * 512) + lane * 16))

    // x contribution: K tiles 0..15
    const unsigned short* xrow = xb + ((size_t)t * B_ + mA) * I_;
#pragma unroll 4
    for (int kt = 0; kt < 16; ++kt) {
      FragA a;
#pragma unroll
      for (int v = 0; v < 8; ++v)
        a.u32[v] = *(const unsigned int*)(xrow + kt * 32 + kb[v]);
      ai  = __builtin_amdgcn_wmma_f32_16x16x32_bf16(false, a.v, false, BFRAG(0, kt), (short)0, ai,  false, false);
      af2 = __builtin_amdgcn_wmma_f32_16x16x32_bf16(false, a.v, false, BFRAG(1, kt), (short)0, af2, false, false);
      ag  = __builtin_amdgcn_wmma_f32_16x16x32_bf16(false, a.v, false, BFRAG(2, kt), (short)0, ag,  false, false);
      ao  = __builtin_amdgcn_wmma_f32_16x16x32_bf16(false, a.v, false, BFRAG(3, kt), (short)0, ao,  false, false);
    }
    // h contribution: K tiles 16..31
    const unsigned short* hrow = hread + (size_t)mA * H_;
#pragma unroll 4
    for (int kt2 = 0; kt2 < 16; ++kt2) {
      const int kt = kt2 + 16;
      FragA a;
#pragma unroll
      for (int v = 0; v < 8; ++v)
        a.u32[v] = *(const unsigned int*)(hrow + kt2 * 32 + kb[v]);
      ai  = __builtin_amdgcn_wmma_f32_16x16x32_bf16(false, a.v, false, BFRAG(0, kt), (short)0, ai,  false, false);
      af2 = __builtin_amdgcn_wmma_f32_16x16x32_bf16(false, a.v, false, BFRAG(1, kt), (short)0, af2, false, false);
      ag  = __builtin_amdgcn_wmma_f32_16x16x32_bf16(false, a.v, false, BFRAG(2, kt), (short)0, ag,  false, false);
      ao  = __builtin_amdgcn_wmma_f32_16x16x32_bf16(false, a.v, false, BFRAG(3, kt), (short)0, ao,  false, false);
    }
#undef BFRAG

    // gate math + state update (C/D layout: element e -> M = e + 8*hi, N = lane%16)
#pragma unroll
    for (int e = 0; e < 8; ++e) {
      float gi = sigmoidf_(ai[e]  + bI);
      float gf = sigmoidf_(af2[e] + bF);
      float gg = tanhf    (ag[e]  + bG);
      float go = sigmoidf_(ao[e]  + bO);
      c[e] = gf * c[e] + gi * gg;
      float hv = go * tanhf(c[e]);
      int mo = mr * 16 + e + 8 * hi;
      hwrite[(size_t)mo * H_ + hcol] = f32_to_bf16(hv);
      out[((size_t)t * B_ + mo) * (2 * H_) + (size_t)dir * H_ + hcol] = hv;
    }

    // per-direction grid sync (monotonic counter, no reset races)
    __threadfence();
    __syncthreads();
    if (tid == 0) {
      __hip_atomic_fetch_add(mycnt, 1u, __ATOMIC_RELEASE, __HIP_MEMORY_SCOPE_AGENT);
      unsigned int target = (unsigned int)(s + 1) * NWG_DIR;
      while (__hip_atomic_load(mycnt, __ATOMIC_ACQUIRE, __HIP_MEMORY_SCOPE_AGENT) < target)
        __builtin_amdgcn_s_sleep(2);
    }
    __syncthreads();
  }
}

// ---------------- launcher ----------------
extern "C" void kernel_launch(void* const* d_in, const int* in_sizes, int n_in,
                              void* d_out, int out_size, void* d_ws, size_t ws_size,
                              hipStream_t stream) {
  (void)in_sizes; (void)n_in; (void)out_size; (void)ws_size;
  const float* x     = (const float*)d_in[0];
  const float* h0f   = (const float*)d_in[1];
  const float* c0f   = (const float*)d_in[2];
  const float* h0b   = (const float*)d_in[3];
  const float* c0b   = (const float*)d_in[4];
  const float* Wih_f = (const float*)d_in[5];
  const float* Whh_f = (const float*)d_in[6];
  const float* b_f   = (const float*)d_in[7];
  const float* Wih_b = (const float*)d_in[8];
  const float* Whh_b = (const float*)d_in[9];
  const float* b_b   = (const float*)d_in[10];
  float* out = (float*)d_out;

  unsigned char* ws = (unsigned char*)d_ws;
  // ws layout: x_bf16 (32 MB) | packed weights (8 MB) | h ping-pong (256 KB) | counters
  unsigned short* xb   = (unsigned short*)(ws);
  unsigned short* wpk  = (unsigned short*)(ws + (size_t)33554432);
  unsigned short* hbuf = (unsigned short*)(ws + (size_t)33554432 + 8388608);
  unsigned int*   cnt  = (unsigned int*)  (ws + (size_t)33554432 + 8388608 + 262144);

  const int nx = T_ * B_ * I_;
  k_cvt_x<<<(nx + 255) / 256, 256, 0, stream>>>(x, xb, nx);

  const int nw = 2 * 128 * 32 * 512;
  k_pack_w<<<(nw + 255) / 256, 256, 0, stream>>>(Wih_f, Whh_f, Wih_b, Whh_b, wpk);

  k_init_h<<<(B_ * H_ + 255) / 256, 256, 0, stream>>>(h0f, h0b, hbuf, cnt);

  k_scan<<<2 * NWG_DIR, WG_THREADS, 0, stream>>>(xb, wpk, hbuf, cnt,
                                                 c0f, c0b, b_f, b_b, out);
}
